// CausalSelfAttention_27084063768738
// MI455X (gfx1250) — matrix operand order repaired and checked by reading, not compiled, not run on hardware
//
#include <hip/hip_runtime.h>
#include <hip/hip_bf16.h>

// ---------------------------------------------------------------------------
// CDNA5 (gfx1250) causal self-attention, bf16 WMMA with fp32 accumulation.
//   - QKV / proj GEMMs: 64x64 tile per wave (16 WMMA accumulators)
//   - attention: flash-style, 32 queries per wave, K/V frags shared by 2 tiles
// ---------------------------------------------------------------------------

typedef __bf16 bf16;
typedef __bf16 v16bf __attribute__((ext_vector_type(16)));
typedef float  v8f   __attribute__((ext_vector_type(8)));

#define DEV __device__ __forceinline__

static constexpr int Tq = 4096;   // sequence length
static constexpr int Cc = 1024;   // channels
static constexpr int Hh = 8;      // heads
static constexpr int Dd = 128;    // head dim

union ABFrag { v16bf v; uint4 u[2]; };

// A operand (16x32 bf16, M x K), memory row-major, leading dim lda.
// Lanes 0-15: M=lane, K={0..7,16..23}; lanes 16-31: same rows, K={8..15,24..31}.
DEV v16bf load_a_frag(const bf16* __restrict__ base, int lda, int m0, int k0, int lane) {
  const int h   = lane >> 4;
  const int row = lane & 15;
  const bf16* p = base + (size_t)(m0 + row) * lda + k0 + 8 * h;
  ABFrag f;
  f.u[0] = *reinterpret_cast<const uint4*>(p);
  f.u[1] = *reinterpret_cast<const uint4*>(p + 16);
  return f.v;
}

// B operand (32x16 bf16, K x N) from W[n][k] row-major (B = W^T), lead dim ldb.
// Lanes 0-15: N=lane, K={0..7,16..23}; lanes 16-31: same cols, K={8..15,24..31}.
DEV v16bf load_b_frag(const bf16* __restrict__ base, int ldb, int n0, int k0, int lane) {
  const int h   = lane >> 4;
  const int col = lane & 15;
  const bf16* p = base + (size_t)(n0 + col) * ldb + k0 + 8 * h;
  ABFrag f;
  f.u[0] = *reinterpret_cast<const uint4*>(p);
  f.u[1] = *reinterpret_cast<const uint4*>(p + 16);
  return f.v;
}

DEV v8f wmma_bf16(v16bf a, v16bf b, v8f c) {
  return __builtin_amdgcn_wmma_f32_16x16x32_bf16(false, a, false, b, (short)0, c, false, false);
}

// ---------------------------------------------------------------------------
// fp32 -> bf16 elementwise convert
// ---------------------------------------------------------------------------
__global__ void cvt_f32_bf16_kernel(const float* __restrict__ src,
                                    bf16* __restrict__ dst, int n) {
  int i = blockIdx.x * blockDim.x + threadIdx.x;
  if (i < n) dst[i] = (bf16)src[i];
}

// ---------------------------------------------------------------------------
// QKV projection: qkv[m,n] = sum_k x[m,k] * w_attn[n,k] + b_attn[n]
// Wave computes a 64x64 tile (4x4 WMMA accumulators). Block = 8 waves.
// Scatter: Q (pre-scaled by 1/sqrt(D)) -> [H,T,D], K -> [H,T,D], V^T -> [H,D,T]
// ---------------------------------------------------------------------------
__global__ __launch_bounds__(256) void qkv_gemm_kernel(
    const bf16* __restrict__ xb, const bf16* __restrict__ wb,
    const float* __restrict__ b_attn,
    bf16* __restrict__ qb, bf16* __restrict__ kb, bf16* __restrict__ vtb) {
  const int lane = threadIdx.x & 31;
  const int wid  = blockIdx.x * 8 + (threadIdx.x >> 5);
  const int TN   = (3 * Cc) / 64;               // 48 tiles along N
  const int m0   = (wid / TN) * 64;
  const int n0   = (wid % TN) * 64;

  v8f acc[4][4] = {};
  for (int k0 = 0; k0 < Cc; k0 += 32) {
    v16bf a[4], b[4];
#pragma unroll
    for (int i = 0; i < 4; ++i) a[i] = load_a_frag(xb, Cc, m0 + 16 * i, k0, lane);
#pragma unroll
    for (int j = 0; j < 4; ++j) b[j] = load_b_frag(wb, Cc, n0 + 16 * j, k0, lane);
    if (k0 + 32 < Cc) {   // gfx1250 prefetch of next k-tile
      __builtin_prefetch(xb + (size_t)(m0 + (lane & 15)) * Cc + k0 + 32, 0, 1);
      __builtin_prefetch(wb + (size_t)(n0 + (lane & 15)) * Cc + k0 + 32, 0, 1);
    }
#pragma unroll
    for (int i = 0; i < 4; ++i)
#pragma unroll
      for (int j = 0; j < 4; ++j)
        acc[i][j] = wmma_bf16(a[i], b[j], acc[i][j]);
  }

  const float qscale = 0.08838834764831845f;    // 1/sqrt(128)
  const int half = lane >> 4;
  const int col  = lane & 15;
#pragma unroll
  for (int j = 0; j < 4; ++j) {
    const int ncol  = n0 + 16 * j + col;
    const float bia = b_attn[ncol];
    const int which = ncol / Cc;                // uniform over 16-wide tile
    const int rr    = ncol % Cc;
    const int h     = rr / Dd;
    const int d     = rr % Dd;
#pragma unroll
    for (int i = 0; i < 4; ++i)
#pragma unroll
      for (int r = 0; r < 8; ++r) {
        const int row = m0 + 16 * i + r + 8 * half;
        const float val = acc[i][j][r] + bia;
        if (which == 0)
          qb[((size_t)h * Tq + row) * Dd + d] = (bf16)(val * qscale);
        else if (which == 1)
          kb[((size_t)h * Tq + row) * Dd + d] = (bf16)val;
        else
          vtb[((size_t)h * Dd + d) * Tq + row] = (bf16)val;   // transposed V
      }
  }
}

// ---------------------------------------------------------------------------
// Flash-style causal attention. One wave handles 32 query rows (two 16-row
// WMMA tiles that share every K and V fragment). Block = 4 waves.
// ---------------------------------------------------------------------------
__global__ __launch_bounds__(128) void attn_kernel(
    const bf16* __restrict__ qb, const bf16* __restrict__ kb,
    const bf16* __restrict__ vtb, bf16* __restrict__ ob) {
  __shared__ alignas(16) bf16 plds[4][2][16 * 32];   // 8 KB

  const int lane = threadIdx.x & 31;
  const int w    = threadIdx.x >> 5;
  const int h    = blockIdx.y;
  const int q0   = (blockIdx.x * 4 + w) * 32;        // 32 queries per wave

  const bf16* Q  = qb  + (size_t)h * Tq * Dd;
  const bf16* K  = kb  + (size_t)h * Tq * Dd;
  const bf16* Vt = vtb + (size_t)h * Dd * Tq;

  const int half = lane >> 4;
  const int col  = lane & 15;
  const float NEG_INF = -__builtin_inff();

  v16bf qfrag[2][4];
#pragma unroll
  for (int qi = 0; qi < 2; ++qi)
#pragma unroll
    for (int kk = 0; kk < 4; ++kk)
      qfrag[qi][kk] = load_a_frag(Q, Dd, q0 + 16 * qi, kk * 32, lane);

  v8f o[2][8] = {};
  float m[2][8], l[2][8];
#pragma unroll
  for (int qi = 0; qi < 2; ++qi)
#pragma unroll
    for (int r = 0; r < 8; ++r) { m[qi][r] = NEG_INF; l[qi][r] = 0.0f; }

  for (int s0 = 0; s0 < q0 + 32; s0 += 32) {
    // ---- S = Q K^T for 32 keys; K frags shared by both query tiles ----
    v8f st[2][2];
#pragma unroll
    for (int t2 = 0; t2 < 2; ++t2) {
      const int ks = s0 + 16 * t2;
      v16bf kf[4];
#pragma unroll
      for (int kk = 0; kk < 4; ++kk)
        kf[kk] = load_b_frag(K, Dd, ks, kk * 32, lane);
      const int key = ks + col;
#pragma unroll
      for (int qi = 0; qi < 2; ++qi) {
        v8f s = {};
#pragma unroll
        for (int kk = 0; kk < 4; ++kk)
          s = wmma_bf16(qfrag[qi][kk], kf[kk], s);
#pragma unroll
        for (int r = 0; r < 8; ++r) {
          const int row = q0 + 16 * qi + r + 8 * half;
          s[r] = (key > row) ? NEG_INF : s[r];   // branch-free causal mask
        }
        st[qi][t2] = s;
      }
    }

    // ---- per-tile online softmax + P -> LDS (C-layout -> A-layout) ----
#pragma unroll
    for (int qi = 0; qi < 2; ++qi) {
      float alpha[8];
#pragma unroll
      for (int r = 0; r < 8; ++r) {
        float mx = fmaxf(st[qi][0][r], st[qi][1][r]);
#pragma unroll
        for (int off = 8; off >= 1; off >>= 1)
          mx = fmaxf(mx, __shfl_xor(mx, off, 32));
        const float mnew = fmaxf(m[qi][r], mx);   // fully-masked block: mnew=m
        alpha[r] = __expf(m[qi][r] - mnew);
        m[qi][r] = mnew;
      }
      bf16* pl = plds[w][qi];
#pragma unroll
      for (int r = 0; r < 8; ++r) {
        const float e0 = __expf(st[qi][0][r] - m[qi][r]);
        const float e1 = __expf(st[qi][1][r] - m[qi][r]);
        float rs = e0 + e1;
#pragma unroll
        for (int off = 8; off >= 1; off >>= 1)
          rs += __shfl_xor(rs, off, 32);
        l[qi][r] = l[qi][r] * alpha[r] + rs;
        const int row = r + 8 * half;
        pl[row * 32 + col]      = (bf16)e0;
        pl[row * 32 + 16 + col] = (bf16)e1;
      }
#pragma unroll
      for (int j = 0; j < 8; ++j)
#pragma unroll
        for (int r = 0; r < 8; ++r)
          o[qi][j][r] *= alpha[r];
    }

    // ---- read both P tiles in A layout ----
    v16bf pa[2];
#pragma unroll
    for (int qi = 0; qi < 2; ++qi) {
      const bf16* pp = plds[w][qi] + col * 32 + 8 * half;
      ABFrag f;
      f.u[0] = *reinterpret_cast<const uint4*>(pp);
      f.u[1] = *reinterpret_cast<const uint4*>(pp + 16);
      pa[qi] = f.v;
    }

    // ---- O += P V : each V fragment feeds both query tiles ----
#pragma unroll
    for (int j = 0; j < 8; ++j) {
      v16bf vf = load_b_frag(Vt, Tq, j * 16, s0, lane);
      o[0][j] = wmma_bf16(pa[0], vf, o[0][j]);
      o[1][j] = wmma_bf16(pa[1], vf, o[1][j]);
    }
  }

  // ---- normalize and emit O as bf16 [T, C] (col = h*D + d) ----
#pragma unroll
  for (int qi = 0; qi < 2; ++qi)
#pragma unroll
    for (int r = 0; r < 8; ++r) {
      const float inv = 1.0f / l[qi][r];
      const int row = q0 + 16 * qi + r + 8 * half;
#pragma unroll
      for (int j = 0; j < 8; ++j) {
        const int c = h * Dd + j * 16 + col;
        ob[(size_t)row * Cc + c] = (bf16)(o[qi][j][r] * inv);
      }
    }
}

// ---------------------------------------------------------------------------
// Output projection: y[m,n] = sum_k O[m,k] * w_proj[n,k] + b_proj[n], fp32 out
// Wave computes a 64x64 tile.
// ---------------------------------------------------------------------------
__global__ __launch_bounds__(256) void proj_gemm_kernel(
    const bf16* __restrict__ ob, const bf16* __restrict__ wpb,
    const float* __restrict__ b_proj, float* __restrict__ y) {
  const int lane = threadIdx.x & 31;
  const int wid  = blockIdx.x * 8 + (threadIdx.x >> 5);
  const int TN   = Cc / 64;                     // 16 tiles along N
  const int m0   = (wid / TN) * 64;
  const int n0   = (wid % TN) * 64;

  v8f acc[4][4] = {};
  for (int k0 = 0; k0 < Cc; k0 += 32) {
    v16bf a[4], b[4];
#pragma unroll
    for (int i = 0; i < 4; ++i) a[i] = load_a_frag(ob, Cc, m0 + 16 * i, k0, lane);
#pragma unroll
    for (int j = 0; j < 4; ++j) b[j] = load_b_frag(wpb, Cc, n0 + 16 * j, k0, lane);
    if (k0 + 32 < Cc) {
      __builtin_prefetch(ob + (size_t)(m0 + (lane & 15)) * Cc + k0 + 32, 0, 1);
      __builtin_prefetch(wpb + (size_t)(n0 + (lane & 15)) * Cc + k0 + 32, 0, 1);
    }
#pragma unroll
    for (int i = 0; i < 4; ++i)
#pragma unroll
      for (int j = 0; j < 4; ++j)
        acc[i][j] = wmma_bf16(a[i], b[j], acc[i][j]);
  }

  const int half = lane >> 4;
  const int col  = lane & 15;
#pragma unroll
  for (int j = 0; j < 4; ++j) {
    const int ncol  = n0 + 16 * j + col;
    const float bia = b_proj[ncol];
#pragma unroll
    for (int i = 0; i < 4; ++i)
#pragma unroll
      for (int r = 0; r < 8; ++r) {
        const int row = m0 + 16 * i + r + 8 * half;
        y[(size_t)row * Cc + ncol] = acc[i][j][r] + bia;
      }
  }
}

// ---------------------------------------------------------------------------
// Host launcher
// ---------------------------------------------------------------------------
extern "C" void kernel_launch(void* const* d_in, const int* in_sizes, int n_in,
                              void* d_out, int out_size, void* d_ws, size_t ws_size,
                              hipStream_t stream) {
  const float* x      = (const float*)d_in[0];
  const float* w_attn = (const float*)d_in[1];
  const float* b_attn = (const float*)d_in[2];
  const float* w_proj = (const float*)d_in[3];
  const float* b_proj = (const float*)d_in[4];
  float* out = (float*)d_out;

  char* ws = (char*)d_ws;
  size_t off = 0;
  auto walloc = [&](size_t bytes) -> void* {
    void* p = ws + off;
    off = (off + bytes + 255) & ~(size_t)255;
    return p;
  };
  bf16* xb   = (bf16*)walloc((size_t)Tq * Cc * 2);        // x in bf16
  bf16* wb   = (bf16*)walloc((size_t)3 * Cc * Cc * 2);    // w_attn bf16
  bf16* wpb  = (bf16*)walloc((size_t)Cc * Cc * 2);        // w_proj bf16
  bf16* qb   = (bf16*)walloc((size_t)Hh * Tq * Dd * 2);   // Q [H,T,D]
  bf16* kb   = (bf16*)walloc((size_t)Hh * Tq * Dd * 2);   // K [H,T,D]
  bf16* vtb  = (bf16*)walloc((size_t)Hh * Dd * Tq * 2);   // V^T [H,D,T]
  bf16* obuf = (bf16*)walloc((size_t)Tq * Cc * 2);        // attn out [T,C]

  {
    int n = Tq * Cc;
    cvt_f32_bf16_kernel<<<(n + 255) / 256, 256, 0, stream>>>(x, xb, n);
  }
  {
    int n = 3 * Cc * Cc;
    cvt_f32_bf16_kernel<<<(n + 255) / 256, 256, 0, stream>>>(w_attn, wb, n);
  }
  {
    int n = Cc * Cc;
    cvt_f32_bf16_kernel<<<(n + 255) / 256, 256, 0, stream>>>(w_proj, wpb, n);
  }

  // QKV GEMM: (4096/64)*(3072/64) = 3072 wave-tiles, 8 waves/block
  qkv_gemm_kernel<<<3072 / 8, 256, 0, stream>>>(xb, wb, b_attn, qb, kb, vtb);

  // Attention: (T/32) query blocks x H heads, 4 waves per block
  attn_kernel<<<dim3(Tq / 32 / 4, Hh), 128, 0, stream>>>(qb, kb, vtb, obuf);

  // Output projection: (4096/64)*(1024/64) = 1024 wave-tiles, 8 waves/block
  proj_gemm_kernel<<<1024 / 8, 256, 0, stream>>>(obuf, wpb, b_proj, out);
}
